// set2set_59021440582056
// MI455X (gfx1250) — matrix operand order repaired and checked by reading, not compiled, not run on hardware
//
#include <hip/hip_runtime.h>
#include <hip/hip_bf16.h>
#include <math.h>

#define BGRAPHS 32768
#define FDIM    128
#define NODES   32      // nodes per graph (N/B)
#define KDIM    256     // folded: gates = q_star @ [W_ih[:,:128]+W_hh | W_ih[:,128:]]^T
#define GATES   512     // 4F
#define STEPS   6

typedef __attribute__((ext_vector_type(16))) __bf16 v16bf;
typedef __attribute__((ext_vector_type(8)))  float  v8f;

union AB16 { v16bf v; uint4 h[2]; };

__device__ __forceinline__ unsigned short f2bf(float f) {
    unsigned u = __float_as_uint(f);
    u += 0x7FFFu + ((u >> 16) & 1u);           // round-to-nearest-even
    return (unsigned short)(u >> 16);
}
__device__ __forceinline__ float sigm(float x) { return 1.0f / (1.0f + __expf(-x)); }

// ---------------------------------------------------------------- zero
__global__ void zero_kernel(float* __restrict__ p, int n) {
    int i = blockIdx.x * blockDim.x + threadIdx.x;
    if (i < n) p[i] = 0.0f;
}

// ------------------------------------------------- weight prep (fold + bf16)
// Wbf[n][k] = bf16( k<128 ? W_ih[n][k] + W_hh[n][k] : W_ih[n][k] )
// bias[n]   = b_ih[n] + b_hh[n]
__global__ void prep_kernel(const float* __restrict__ W_ih,
                            const float* __restrict__ W_hh,
                            const float* __restrict__ b_ih,
                            const float* __restrict__ b_hh,
                            unsigned short* __restrict__ Wbf,
                            float* __restrict__ bias) {
    int i = blockIdx.x * blockDim.x + threadIdx.x;
    if (i < GATES * KDIM) {
        int n = i >> 8, k = i & (KDIM - 1);
        float w = W_ih[n * KDIM + k];
        if (k < FDIM) w += W_hh[n * FDIM + k];
        Wbf[i] = f2bf(w);
    }
    if (i < GATES) bias[i] = b_ih[i] + b_hh[i];
}

// ------------------------------------------------- fused LSTM cell via WMMA
// Per block: 16 rows (graphs) x all 512 gate columns.
// gates = Abf(q_star in bf16) @ Wbf^T  (8 k-steps x 4 WMMA/wave, f32 acc),
// gate exchange through 32KB LDS, fused sigmoid/tanh state update.
// gates never touch HBM; A is pre-converted bf16 (no VALU convert in loop).
__global__ __launch_bounds__(256)
void lstm_gemm_kernel(float* __restrict__ qstar,               // [B,256] (cols 0..127 written)
                      float* __restrict__ cbuf,                // [B,128]
                      unsigned short* __restrict__ Abf,        // [B,256] bf16 mirror of q_star
                      const unsigned short* __restrict__ Wbf,  // [512][256] bf16
                      const float* __restrict__ bias) {        // [512]
    __shared__ float gl[16 * GATES];                           // 32 KB gate exchange

    const int tid  = threadIdx.x;
    const int wave = tid >> 5;
    const int lane = tid & 31;
    const int half = lane >> 4;
    const int r    = lane & 15;
    const int m0   = blockIdx.x * 16;
    const int row  = m0 + r;

    v8f z = {0,0,0,0,0,0,0,0};
    v8f acc[4] = {z, z, z, z};

    const unsigned short* arow = Abf + (size_t)row * KDIM;

#pragma unroll 2
    for (int kt = 0; kt < KDIM / 32; ++kt) {
        const int k0 = kt * 32;
        // A fragment: 16x32 bf16, contiguous K (half selects K 0-15 vs 16-31)
        AB16 af;
        af.h[0] = *(const uint4*)(arow + k0 + half * 16);
        af.h[1] = *(const uint4*)(arow + k0 + half * 16 + 8);
        // B fragments first (clause the loads), then the 4 WMMAs back-to-back
        v16bf bf[4];
#pragma unroll
        for (int t = 0; t < 4; ++t) {
            const unsigned short* wrow =
                Wbf + (size_t)((((wave << 2) + t) << 4) + r) * KDIM;
            bf[t] = *(const v16bf*)(wrow + k0 + half * 16);
        }
#pragma unroll
        for (int t = 0; t < 4; ++t)
            acc[t] = __builtin_amdgcn_wmma_f32_16x16x32_bf16(
                false, af.v, false, bf[t], (short)0, acc[t], false, false);
    }

    // D layout: M = v + 8*half, N = lane&15 -> scatter into LDS gate matrix
#pragma unroll
    for (int t = 0; t < 4; ++t) {
        const int ncol = (((wave << 2) + t) << 4) + r;
#pragma unroll
        for (int v = 0; v < 8; ++v)
            gl[(v + 8 * half) * GATES + ncol] = acc[t][v];
    }
    __syncthreads();

    // fused LSTM elementwise: 16 rows x 128 features
    for (int i = tid; i < 16 * FDIM; i += 256) {
        const int m = i >> 7;
        const int j = i & (FDIM - 1);
        const int rr = m0 + m;
        const float gi = gl[m * GATES + j]            + bias[j];
        const float gf = gl[m * GATES + FDIM + j]     + bias[FDIM + j];
        const float gg = gl[m * GATES + 2 * FDIM + j] + bias[2 * FDIM + j];
        const float go = gl[m * GATES + 3 * FDIM + j] + bias[3 * FDIM + j];
        const float co = cbuf[(size_t)rr * FDIM + j];
        const float cn = sigm(gf) * co + sigm(gi) * tanhf(gg);
        const float hn = sigm(go) * tanhf(cn);
        cbuf[(size_t)rr * FDIM + j] = cn;
        qstar[(size_t)rr * (2 * FDIM) + j] = hn;       // q -> q_star[:, :128] (fp32)
        Abf[(size_t)rr * KDIM + j] = f2bf(hn);         // q -> bf16 A mirror (one convert)
    }
}

// ------------------------------------------------- attention readout
// 1 wave per graph (2 graphs / 64-thread block). 32x128 tile staged once in
// LDS (rows padded to 132 floats); logits, wave32-shuffle softmax, weighted
// sum all hit the LDS tile -> x streams from HBM exactly once per step.
__global__ __launch_bounds__(64)
void attn_kernel(const float* __restrict__ x,
                 float* __restrict__ qstar,
                 unsigned short* __restrict__ Abf) {
    __shared__ float tile[2][NODES][FDIM + 4];
    __shared__ float qs[2][FDIM];

    const int wave = threadIdx.x >> 5;
    const int lane = threadIdx.x & 31;
    const int g    = blockIdx.x * 2 + wave;
    const float* xg  = x + (size_t)g * NODES * FDIM;
    float*       qsg = qstar + (size_t)g * (2 * FDIM);

    ((float4*)qs[wave])[lane] = ((const float4*)qsg)[lane];   // q broadcast copy

    // phase 1: coalesced tile load (32 lanes x 16B = one 512B row per iter)
    for (int n = 0; n < NODES; ++n) {
        float4 xv = ((const float4*)(xg + n * FDIM))[lane];
        *(float4*)&tile[wave][n][4 * lane] = xv;
    }
    __syncthreads();

    // phase 2: lane n computes e_n = dot(x_row_n, q) from LDS
    float e = 0.0f;
    const float* rowp = tile[wave][lane];
    for (int c = 0; c < FDIM / 4; ++c) {
        const float4 xv = *(const float4*)&rowp[4 * c];
        const float4 qq = ((const float4*)qs[wave])[c];
        e += xv.x * qq.x + xv.y * qq.y + xv.z * qq.z + xv.w * qq.w;
    }

    // stable softmax across the 32 lanes
    float mx = e;
    for (int s = 16; s > 0; s >>= 1) mx = fmaxf(mx, __shfl_xor(mx, s, 32));
    float a = __expf(e - mx);
    float sum = a;
    for (int s = 16; s > 0; s >>= 1) sum += __shfl_xor(sum, s, 32);
    a /= sum;

    // phase 3: r[4*lane..] = sum_n a_n * x[n][4*lane..]
    float4 accv = {0.0f, 0.0f, 0.0f, 0.0f};
    for (int n = 0; n < NODES; ++n) {
        const float an = __shfl(a, n, 32);
        const float4 xv = *(const float4*)&tile[wave][n][4 * lane];
        accv.x += an * xv.x; accv.y += an * xv.y;
        accv.z += an * xv.z; accv.w += an * xv.w;
    }
    ((float4*)(qsg + FDIM))[lane] = accv;              // r -> q_star[:, 128:] (fp32)

    ushort4 rb;                                        // r -> bf16 A mirror (one convert)
    rb.x = f2bf(accv.x); rb.y = f2bf(accv.y);
    rb.z = f2bf(accv.z); rb.w = f2bf(accv.w);
    *(ushort4*)(Abf + (size_t)g * KDIM + FDIM + 4 * lane) = rb;
}

// ---------------------------------------------------------------- launch
extern "C" void kernel_launch(void* const* d_in, const int* in_sizes, int n_in,
                              void* d_out, int out_size, void* d_ws, size_t ws_size,
                              hipStream_t stream) {
    const float* x     = (const float*)d_in[0];
    // d_in[1] = batch: uniform 32 consecutive nodes per graph -> index math
    const float* W_ih  = (const float*)d_in[2];
    const float* W_hh  = (const float*)d_in[3];
    const float* b_ih  = (const float*)d_in[4];
    const float* b_hh  = (const float*)d_in[5];

    float* qstar = (float*)d_out;                        // [B, 2F], final output
    float* cbuf  = (float*)d_ws;                         // [B, F] fp32 cell state
    unsigned short* Abf = (unsigned short*)(cbuf + (size_t)BGRAPHS * FDIM); // [B,256] bf16
    unsigned short* Wbf = Abf + (size_t)BGRAPHS * KDIM;                     // [512][256] bf16
    float* bias  = (float*)(Wbf + (size_t)GATES * KDIM);                    // [512]

    // c (fp32) and Abf (bf16) are contiguous: zero both in one sweep
    const int nCA = BGRAPHS * FDIM + (BGRAPHS * KDIM) / 2;  // floats
    const int nQ  = BGRAPHS * 2 * FDIM;
    zero_kernel<<<(nCA + 255) / 256, 256, 0, stream>>>(cbuf, nCA);
    zero_kernel<<<(nQ + 255) / 256, 256, 0, stream>>>(qstar, nQ);
    prep_kernel<<<(GATES * KDIM + 255) / 256, 256, 0, stream>>>(
        W_ih, W_hh, b_ih, b_hh, Wbf, bias);

    for (int step = 0; step < STEPS; ++step) {
        lstm_gemm_kernel<<<BGRAPHS / 16, 256, 0, stream>>>(qstar, cbuf, Abf, Wbf, bias);
        attn_kernel<<<BGRAPHS / 2, 64, 0, stream>>>(x, qstar, Abf);
    }
}
